// PUNet_7009386627690
// MI455X (gfx1250) — compile-verified, matrix operand order checked by reading, not run on hardware
//
#include <hip/hip_runtime.h>
#include <stdint.h>

// ---------------------------------------------------------------------------
// PU-Net forward for MI455X (gfx1250). All MLP/1x1-conv layers run as
// bf16 WMMA GEMMs (v_wmma_f32_16x16x32_bf16), f32 accumulate, with the
// weight strip staged in LDS (async-to-LDS when available) and shared by
// all 8 waves of the workgroup.
// ---------------------------------------------------------------------------

typedef __attribute__((ext_vector_type(16))) __bf16 v16bf;
typedef __attribute__((ext_vector_type(8)))  float  v8f;
typedef int v4i __attribute__((vector_size(16)));

#if __has_builtin(__builtin_amdgcn_global_load_async_to_lds_b128) && \
    __has_builtin(__builtin_amdgcn_s_wait_asynccnt)
#define ASYNC_LDS 1
#endif

__device__ __forceinline__ unsigned short f2bf(float f) {
  unsigned int u = __float_as_uint(f);
  u += 0x7FFFu + ((u >> 16) & 1u);          // round-to-nearest-even
  return (unsigned short)(u >> 16);
}
__device__ __forceinline__ float bf2f(unsigned short h) {
  return __uint_as_float(((unsigned int)h) << 16);
}

union FragAB { v16bf v; uint4 q[2]; };
union FragC  { v8f  v; float f[8]; };

// ---------------------------------------------------------------------------
// WMMA GEMM:  C[M,N] = relu(A[M,Kp] * W[N,Kp]^T + bias[N]),  bf16 in/out.
// Kp % 32 == 0, M % 128 == 0, N % (NT*16) == 0.
// Block = 256 threads = 8 waves; block tile = 128 rows x NT*16 cols.
// W strip for one 32-wide K-chunk is double-buffered in LDS.
// ---------------------------------------------------------------------------
#define LWPAD 40  // 32 data shorts + 8 pad -> 80B row stride (bank spread)

__device__ __forceinline__ void stage_chunk(const unsigned short* __restrict__ W,
                                            unsigned short* lbuf,
                                            int n0, int Kp, int k0,
                                            int nsegs, int tid) {
  // copy cols x 32 shorts: segment = 8 shorts (16B); nsegs = cols*4
  for (int seg = tid; seg < nsegs; seg += 256) {
    const int row = seg >> 2, q = seg & 3;
    const unsigned short* g = W + (size_t)(n0 + row) * Kp + k0 + q * 8;
    unsigned short* l = lbuf + row * LWPAD + q * 8;
#ifdef ASYNC_LDS
    __builtin_amdgcn_global_load_async_to_lds_b128(
        (__attribute__((address_space(1))) v4i*)g,
        (__attribute__((address_space(3))) v4i*)l, 0, 0);
#else
    *(uint4*)l = *(const uint4*)g;
#endif
  }
}

__device__ __forceinline__ void wait_stage() {
#ifdef ASYNC_LDS
  __builtin_amdgcn_s_wait_asynccnt(0);
#endif
  __syncthreads();
}

template <int NT>
__global__ __launch_bounds__(256)
void gemm_bias_relu_wmma(const unsigned short* __restrict__ A,
                         const unsigned short* __restrict__ W,
                         const float* __restrict__ bias,
                         unsigned short* __restrict__ C,
                         int M, int Kp, int Nout) {
  __shared__ unsigned short lw[2][128 * LWPAD];
  const int tid  = threadIdx.x;
  const int lane = tid & 31;
  const int wv   = tid >> 5;
  const int l    = lane & 15;
  const int hi   = (lane >> 4) & 1;
  const int koff0 = hi ? 8 : 0, koff1 = hi ? 24 : 16;
  const int n0    = blockIdx.y * (NT * 16);
  const int nsegs = NT * 16 * 4;
  const int nchunks = Kp >> 5;

  float bv[NT];
  int colj[NT];
#pragma unroll
  for (int j = 0; j < NT; ++j) {
    colj[j] = n0 + 16 * j + l;
    bv[j] = bias[colj[j]];
  }

  for (int mc = blockIdx.x; mc < (M >> 7); mc += gridDim.x) {
    const int m0 = (mc << 7) + (wv << 4);
    FragC acc[NT];
#pragma unroll
    for (int j = 0; j < NT; ++j)
#pragma unroll
      for (int r = 0; r < 8; ++r) acc[j].f[r] = 0.0f;

    const unsigned short* arow = A + (size_t)(m0 + l) * Kp;

    stage_chunk(W, lw[0], n0, Kp, 0, nsegs, tid);
    for (int c = 0; c < nchunks; ++c) {
      wait_stage();                       // LDS buf[c&1] ready for all waves
      if (c + 1 < nchunks)
        stage_chunk(W, lw[(c + 1) & 1], n0, Kp, (c + 1) << 5, nsegs, tid);
      const int k0 = c << 5;
      FragAB a;
      a.q[0] = *(const uint4*)(arow + k0 + koff0);
      a.q[1] = *(const uint4*)(arow + k0 + koff1);
      const unsigned short* lbase = &lw[c & 1][0];
#pragma unroll
      for (int j = 0; j < NT; ++j) {
        FragAB b;
        const unsigned short* lr = lbase + (16 * j + l) * LWPAD;
        b.q[0] = *(const uint4*)(lr + koff0);
        b.q[1] = *(const uint4*)(lr + koff1);
        acc[j].v = __builtin_amdgcn_wmma_f32_16x16x32_bf16(
            false, a.v, false, b.v, (short)0, acc[j].v, false, false);
      }
      __syncthreads();                    // reads done before buf restaged
    }

    const int rbase = m0 + (hi ? 8 : 0);
#pragma unroll
    for (int j = 0; j < NT; ++j)
#pragma unroll
      for (int r = 0; r < 8; ++r) {
        float v = fmaxf(acc[j].f[r] + bv[j], 0.0f);
        C[(size_t)(rbase + r) * Nout + colj[j]] = f2bf(v);
      }
  }
}

// ---------------------------------------------------------------------------
// Weight pad+convert: Wp[o, k] = k < Cin ? bf16(W[o, k]) : 0   (Kp stride)
// ---------------------------------------------------------------------------
__global__ void pad_weights_kernel(const float* __restrict__ W,
                                   unsigned short* __restrict__ Wp,
                                   int Cin, int Kp, int total) {
  int idx = blockIdx.x * blockDim.x + threadIdx.x;
  if (idx >= total) return;
  int k = idx % Kp, o = idx / Kp;
  Wp[idx] = (k < Cin) ? f2bf(W[(size_t)o * Cin + k]) : (unsigned short)0;
}

// ---------------------------------------------------------------------------
// Farthest point sampling: one block per batch, sequential scan with block
// argmax reduction in LDS.
// ---------------------------------------------------------------------------
#define FPS_BLOCK 256
__global__ void fps_kernel(const float* __restrict__ xyz, int* __restrict__ fidx,
                           int Nsrc, int npoint) {
  const int b = blockIdx.x;
  const float* px = xyz + (size_t)b * Nsrc * 3;
  __shared__ float dist[1024];
  __shared__ float rmax[FPS_BLOCK];
  __shared__ int   rarg[FPS_BLOCK];
  __shared__ int   s_far;
  for (int i = threadIdx.x; i < Nsrc; i += blockDim.x) dist[i] = 1e10f;
  if (threadIdx.x == 0) s_far = 0;
  __syncthreads();
  for (int step = 0; step < npoint; ++step) {
    const int far = s_far;
    if (threadIdx.x == 0) fidx[(size_t)b * npoint + step] = far;
    const float cx = px[far * 3], cy = px[far * 3 + 1], cz = px[far * 3 + 2];
    float best = -1.0f; int bi = 0;
    for (int i = threadIdx.x; i < Nsrc; i += blockDim.x) {
      float dx = px[i * 3] - cx, dy = px[i * 3 + 1] - cy, dz = px[i * 3 + 2] - cz;
      float nd = fminf(dist[i], dx * dx + dy * dy + dz * dz);
      dist[i] = nd;
      if (nd > best) { best = nd; bi = i; }
    }
    rmax[threadIdx.x] = best; rarg[threadIdx.x] = bi;
    __syncthreads();
    for (int s = FPS_BLOCK / 2; s > 0; s >>= 1) {
      if (threadIdx.x < s) {
        if (rmax[threadIdx.x + s] > rmax[threadIdx.x] ||
            (rmax[threadIdx.x + s] == rmax[threadIdx.x] &&
             rarg[threadIdx.x + s] < rarg[threadIdx.x])) {
          rmax[threadIdx.x] = rmax[threadIdx.x + s];
          rarg[threadIdx.x] = rarg[threadIdx.x + s];
        }
      }
      __syncthreads();
    }
    if (threadIdx.x == 0) s_far = rarg[0];
    __syncthreads();
  }
}

__global__ void gather_xyz2_kernel(const float* __restrict__ xyz,
                                   const int* __restrict__ fidx,
                                   float* __restrict__ nxyz,
                                   int Nsrc, int S, int total) {
  int idx = blockIdx.x * blockDim.x + threadIdx.x;
  if (idx >= total) return;
  int b = idx / S;
  int gi = fidx[idx];
  const float* s = xyz + ((size_t)b * Nsrc + gi) * 3;
  float* d = nxyz + (size_t)idx * 3;
  d[0] = s[0]; d[1] = s[1]; d[2] = s[2];
}

// ---------------------------------------------------------------------------
// Ball query: first K in-range indices in ascending order; pad with first.
// ---------------------------------------------------------------------------
__global__ void ball_query_kernel(const float* __restrict__ xyz,
                                  const float* __restrict__ nxyz,
                                  int* __restrict__ gidx,
                                  int Nsrc, int S, float r2, int K, int total) {
  int idx = blockIdx.x * blockDim.x + threadIdx.x;
  if (idx >= total) return;
  int b = idx / S;
  const float* px = xyz + (size_t)b * Nsrc * 3;
  const float* c = nxyz + (size_t)idx * 3;
  const float cx = c[0], cy = c[1], cz = c[2];
  int* out = gidx + (size_t)idx * K;
  int cnt = 0, first = -1;
  for (int i = 0; i < Nsrc && cnt < K; ++i) {
    float dx = px[i * 3] - cx, dy = px[i * 3 + 1] - cy, dz = px[i * 3 + 2] - cz;
    if (dx * dx + dy * dy + dz * dz < r2) {
      if (first < 0) first = i;
      out[cnt++] = i;
    }
  }
  if (first < 0) first = 0;
  for (int k = cnt; k < K; ++k) out[k] = first;
}

// ---------------------------------------------------------------------------
// Grouping: row (b,s,k) of the SA-MLP input = [xyz[gi]-center, feats[gi]],
// written as bf16 with zero pad out to Kp columns.
// ---------------------------------------------------------------------------
__global__ void group_kernel(const float* __restrict__ xyz,
                             const float* __restrict__ nxyz,
                             const float* __restrict__ feats,
                             const int* __restrict__ gidx,
                             unsigned short* __restrict__ out,
                             int Nsrc, int S, int K, int C, int Kp, int total) {
  int idx = blockIdx.x * blockDim.x + threadIdx.x;
  if (idx >= total) return;
  int bs = idx / K;
  int b = bs / S;
  int gi = gidx[idx];
  const float* p = xyz + ((size_t)b * Nsrc + gi) * 3;
  const float* c = nxyz + (size_t)bs * 3;
  unsigned short* row = out + (size_t)idx * Kp;
  row[0] = f2bf(p[0] - c[0]);
  row[1] = f2bf(p[1] - c[1]);
  row[2] = f2bf(p[2] - c[2]);
  if (C > 0) {
    const float* f = feats + ((size_t)b * Nsrc + gi) * C;
    for (int j = 0; j < C; ++j) row[3 + j] = f2bf(f[j]);
  }
  for (int j = 3 + C; j < Kp; ++j) row[j] = 0;
}

// ---------------------------------------------------------------------------
// Max-pool over nsample dim; bf16 in, f32 out (per (b,s,c)).
// ---------------------------------------------------------------------------
__global__ void maxpool_kernel(const unsigned short* __restrict__ in,
                               float* __restrict__ out,
                               int Ksamp, int C, int total) {
  int idx = blockIdx.x * blockDim.x + threadIdx.x;
  if (idx >= total) return;
  int c = idx % C;
  size_t g = (size_t)(idx / C);
  const unsigned short* p = in + g * (size_t)Ksamp * C + c;
  float m = -3.4e38f;
  for (int k = 0; k < Ksamp; ++k) m = fmaxf(m, bf2f(p[(size_t)k * C]));
  out[idx] = m;
}

// ---------------------------------------------------------------------------
// FP interpolation: 3-NN inverse-distance weighted features -> bf16 rows.
// ---------------------------------------------------------------------------
__global__ void fp_interp_kernel(const float* __restrict__ uxyz,
                                 const float* __restrict__ kxyz,
                                 const float* __restrict__ kfeat,
                                 unsigned short* __restrict__ out,
                                 int Npts, int S, int C, int total) {
  int idx = blockIdx.x * blockDim.x + threadIdx.x;
  if (idx >= total) return;
  int b = idx / Npts;
  const float* u = uxyz + (size_t)idx * 3;
  const float* kx = kxyz + (size_t)b * S * 3;
  const float ux = u[0], uy = u[1], uz = u[2];
  float d0 = 1e30f, d1 = 1e30f, d2 = 1e30f;
  int i0 = 0, i1 = 0, i2 = 0;
  for (int i = 0; i < S; ++i) {
    float dx = kx[i * 3] - ux, dy = kx[i * 3 + 1] - uy, dz = kx[i * 3 + 2] - uz;
    float dd = dx * dx + dy * dy + dz * dz;
    if (dd < d0)      { d2 = d1; i2 = i1; d1 = d0; i1 = i0; d0 = dd; i0 = i; }
    else if (dd < d1) { d2 = d1; i2 = i1; d1 = dd; i1 = i; }
    else if (dd < d2) { d2 = dd; i2 = i; }
  }
  float w0 = 1.0f / (d0 + 1e-8f), w1 = 1.0f / (d1 + 1e-8f), w2 = 1.0f / (d2 + 1e-8f);
  float ws = w0 + w1 + w2; w0 /= ws; w1 /= ws; w2 /= ws;
  const float* f0 = kfeat + ((size_t)b * S + i0) * C;
  const float* f1 = kfeat + ((size_t)b * S + i1) * C;
  const float* f2 = kfeat + ((size_t)b * S + i2) * C;
  unsigned short* row = out + (size_t)idx * C;
  for (int c = 0; c < C; ++c)
    row[c] = f2bf(w0 * f0[c] + w1 * f1[c] + w2 * f2[c]);
}

__global__ void bf16_to_f32_kernel(const unsigned short* __restrict__ in,
                                   float* __restrict__ out, int total) {
  int idx = blockIdx.x * blockDim.x + threadIdx.x;
  if (idx >= total) return;
  out[idx] = bf2f(in[idx]);
}

// ---------------------------------------------------------------------------
// Concat [xyz(3) | l_feats1(64) | up0(64) | up1(64) | up2(64)] -> 259, pad 288.
// ---------------------------------------------------------------------------
__global__ void concat_feats_kernel(const float* __restrict__ xyz,
                                    const float* __restrict__ f1,
                                    const float* __restrict__ u0,
                                    const float* __restrict__ u1,
                                    const float* __restrict__ u2,
                                    unsigned short* __restrict__ out, int total) {
  int idx = blockIdx.x * blockDim.x + threadIdx.x;
  if (idx >= total) return;
  unsigned short* row = out + (size_t)idx * 288;
  const float* x = xyz + (size_t)idx * 3;
  row[0] = f2bf(x[0]); row[1] = f2bf(x[1]); row[2] = f2bf(x[2]);
  const float* a = f1 + (size_t)idx * 64;
  for (int j = 0; j < 64; ++j) row[3 + j] = f2bf(a[j]);
  const float* p0 = u0 + (size_t)idx * 64;
  for (int j = 0; j < 64; ++j) row[67 + j] = f2bf(p0[j]);
  const float* p1 = u1 + (size_t)idx * 64;
  for (int j = 0; j < 64; ++j) row[131 + j] = f2bf(p1[j]);
  const float* p2 = u2 + (size_t)idx * 64;
  for (int j = 0; j < 64; ++j) row[195 + j] = f2bf(p2[j]);
  for (int j = 259; j < 288; ++j) row[j] = 0;
}

// Scatter FC branch output (B,1024,128) into (B,4096,128) at point-offset.
__global__ void scatter_fc_kernel(const unsigned short* __restrict__ src,
                                  unsigned short* __restrict__ dst,
                                  int branch, int total) {
  int idx = blockIdx.x * blockDim.x + threadIdx.x;
  if (idx >= total) return;
  int c = idx & 127;
  int row = idx >> 7;
  int b = row >> 10, n = row & 1023;
  dst[(((size_t)b * 4096) + (size_t)branch * 1024 + n) * 128 + c] = src[idx];
}

// Final 64 -> 3 layer (no relu), f32 output straight to d_out.
__global__ void pcd_final_kernel(const unsigned short* __restrict__ in,
                                 const float* __restrict__ W,   // 3 x 64
                                 const float* __restrict__ bias,  // 3
                                 float* __restrict__ out, int total) {
  int idx = blockIdx.x * blockDim.x + threadIdx.x;
  if (idx >= total) return;
  const unsigned short* r = in + (size_t)idx * 64;
#pragma unroll
  for (int o = 0; o < 3; ++o) {
    float acc = bias[o];
    for (int j = 0; j < 64; ++j) acc += bf2f(r[j]) * W[o * 64 + j];
    out[(size_t)idx * 3 + o] = acc;
  }
}

// ---------------------------------------------------------------------------
// Host orchestration
// ---------------------------------------------------------------------------
static inline int rup32(int x) { return (x + 31) & ~31; }

extern "C" void kernel_launch(void* const* d_in, const int* in_sizes, int n_in,
                              void* d_out, int out_size, void* d_ws, size_t ws_size,
                              hipStream_t stream) {
  (void)in_sizes; (void)n_in; (void)out_size; (void)ws_size;
  const int Bb = 32, Nn = 1024, Ks = 32;
  const int NPTS[4] = {1024, 512, 256, 128};
  const float RAD[4] = {0.05f, 0.1f, 0.2f, 0.3f};
  const int SA_CIN[4][3]  = {{3,32,32},{67,64,64},{131,128,128},{259,256,256}};
  const int SA_COUT[4][3] = {{32,32,64},{64,64,128},{128,128,256},{256,256,512}};
  const int FP_CIN[3] = {128, 256, 512};
  const int FC_CIN[2] = {259, 256};
  const int FC_COUT[2] = {256, 128};

  // ---- input unpack (setup_inputs dict insertion order) ----
  int p = 0;
  const float* points = (const float*)d_in[p++];
  const float *saW[4][3], *saB[4][3];
  for (int i = 0; i < 4; ++i)
    for (int j = 0; j < 3; ++j) { saW[i][j] = (const float*)d_in[p++]; saB[i][j] = (const float*)d_in[p++]; }
  const float *fpW[3], *fpB[3];
  for (int i = 0; i < 3; ++i) { fpW[i] = (const float*)d_in[p++]; fpB[i] = (const float*)d_in[p++]; }
  const float *fcW[4][2], *fcB[4][2];
  for (int i = 0; i < 4; ++i)
    for (int j = 0; j < 2; ++j) { fcW[i][j] = (const float*)d_in[p++]; fcB[i][j] = (const float*)d_in[p++]; }
  const float *pcdW[2], *pcdB[2];
  for (int i = 0; i < 2; ++i) { pcdW[i] = (const float*)d_in[p++]; pcdB[i] = (const float*)d_in[p++]; }

  // ---- workspace bump allocator ----
  uintptr_t cur = (uintptr_t)d_ws;
  auto alloc = [&](size_t bytes) -> void* {
    uintptr_t r = (cur + 255) & ~(uintptr_t)255;
    cur = r + bytes;
    return (void*)r;
  };
  auto cdiv = [](long a, long b) { return (int)((a + b - 1) / b); };

  // padded bf16 weights
  unsigned short *saWp[4][3], *fpWp[3], *fcWp[4][2], *pcdWp;
  auto pad_w = [&](const float* W, int Cout, int Cin) -> unsigned short* {
    int Kp = rup32(Cin);
    int total = Cout * Kp;
    unsigned short* Wp = (unsigned short*)alloc((size_t)total * 2);
    pad_weights_kernel<<<cdiv(total, 256), 256, 0, stream>>>(W, Wp, Cin, Kp, total);
    return Wp;
  };
  for (int i = 0; i < 4; ++i)
    for (int j = 0; j < 3; ++j) saWp[i][j] = pad_w(saW[i][j], SA_COUT[i][j], SA_CIN[i][j]);
  for (int i = 0; i < 3; ++i) fpWp[i] = pad_w(fpW[i], 64, FP_CIN[i]);
  for (int i = 0; i < 4; ++i)
    for (int j = 0; j < 2; ++j) fcWp[i][j] = pad_w(fcW[i][j], FC_COUT[j], FC_CIN[j]);
  pcdWp = pad_w(pcdW[0], 64, 128);

  // level buffers
  int* fidx = (int*)alloc((size_t)Bb * 1024 * sizeof(int));
  int* gidx = (int*)alloc((size_t)Bb * 1024 * Ks * sizeof(int));
  float* lxyz[5];  lxyz[0] = (float*)points;
  float* lfeat[5]; lfeat[0] = nullptr;
  for (int i = 1; i <= 4; ++i) {
    lxyz[i]  = (float*)alloc((size_t)Bb * NPTS[i - 1] * 3 * sizeof(float));
    lfeat[i] = (float*)alloc((size_t)Bb * NPTS[i - 1] * SA_COUT[i - 1][2] * sizeof(float));
  }
  const size_t PPELEMS = 67108864;  // max GEMM operand elems (1M x 64)
  unsigned short* ping = (unsigned short*)alloc(PPELEMS * 2);
  unsigned short* pong = (unsigned short*)alloc(PPELEMS * 2);
  float* up[3];
  for (int i = 0; i < 3; ++i) up[i] = (float*)alloc((size_t)Bb * Nn * 64 * sizeof(float));
  unsigned short* fc_in = (unsigned short*)alloc((size_t)Bb * Nn * 288 * 2);
  unsigned short* rbig  = (unsigned short*)alloc((size_t)Bb * 4096 * 128 * 2);

  // GEMM dispatch: block tile = 128 rows x NT*16 cols
  auto gemm = [&](const unsigned short* A, const unsigned short* Wp, const float* bias,
                  unsigned short* Cb, int M, int Kp, int Nout) {
    int mchunks = M >> 7;
    int blocksM = mchunks < 1024 ? mchunks : 1024;
    if (Nout == 32) {
      dim3 g(blocksM, 1);
      gemm_bias_relu_wmma<2><<<g, 256, 0, stream>>>(A, Wp, bias, Cb, M, Kp, Nout);
    } else if (Nout == 64) {
      dim3 g(blocksM, 1);
      gemm_bias_relu_wmma<4><<<g, 256, 0, stream>>>(A, Wp, bias, Cb, M, Kp, Nout);
    } else {
      dim3 g(blocksM, Nout >> 7);
      gemm_bias_relu_wmma<8><<<g, 256, 0, stream>>>(A, Wp, bias, Cb, M, Kp, Nout);
    }
  };

  // ---- SA modules ----
  int curN = Nn, curC = 0;
  for (int i = 0; i < 4; ++i) {
    const int S = NPTS[i];
    const float r2 = RAD[i] * RAD[i];
    fps_kernel<<<Bb, FPS_BLOCK, 0, stream>>>(lxyz[i], fidx, curN, S);
    int tot_bs = Bb * S;
    gather_xyz2_kernel<<<cdiv(tot_bs, 256), 256, 0, stream>>>(lxyz[i], fidx, lxyz[i + 1], curN, S, tot_bs);
    ball_query_kernel<<<cdiv(tot_bs, 256), 256, 0, stream>>>(lxyz[i], lxyz[i + 1], gidx, curN, S, r2, Ks, tot_bs);
    const int Kp0 = rup32(3 + curC);
    const int rows = Bb * S * Ks;
    group_kernel<<<cdiv(rows, 256), 256, 0, stream>>>(lxyz[i], lxyz[i + 1], lfeat[i], gidx,
                                                      ping, curN, S, Ks, curC, Kp0, rows);
    unsigned short* a = ping; unsigned short* o = pong;
    int kp = Kp0;
    for (int j = 0; j < 3; ++j) {
      gemm(a, saWp[i][j], saB[i][j], o, rows, kp, SA_COUT[i][j]);
      unsigned short* t = a; a = o; o = t;
      kp = SA_COUT[i][j];           // next layer Cin (already mult of 32)
    }
    const int C2 = SA_COUT[i][2];
    const int pool_total = Bb * S * C2;
    maxpool_kernel<<<cdiv(pool_total, 256), 256, 0, stream>>>(a, lfeat[i + 1], Ks, C2, pool_total);
    curN = S; curC = C2;
  }

  // ---- FP modules (interp from levels 2..4 back to original N) ----
  const int BN = Bb * Nn;
  for (int i = 0; i < 3; ++i) {
    const int S = NPTS[i + 1];
    const int C = FP_CIN[i];
    fp_interp_kernel<<<cdiv(BN, 256), 256, 0, stream>>>(points, lxyz[i + 2], lfeat[i + 2],
                                                        ping, Nn, S, C, BN);
    gemm(ping, fpWp[i], fpB[i], pong, BN, C, 64);
    bf16_to_f32_kernel<<<cdiv(BN * 64, 256), 256, 0, stream>>>(pong, up[i], BN * 64);
  }

  // ---- concat -> FC branches -> scatter ----
  concat_feats_kernel<<<cdiv(BN, 256), 256, 0, stream>>>(points, lfeat[1], up[0], up[1], up[2],
                                                         fc_in, BN);
  for (int i = 0; i < 4; ++i) {
    gemm(fc_in, fcWp[i][0], fcB[i][0], pong, BN, 288, 256);
    gemm(pong, fcWp[i][1], fcB[i][1], ping, BN, 256, 128);
    int tot = BN * 128;
    scatter_fc_kernel<<<cdiv(tot, 256), 256, 0, stream>>>(ping, rbig, i, tot);
  }

  // ---- PCD head ----
  const int rowsUp = Bb * 4096;
  gemm(rbig, pcdWp, pcdB[0], ping, rowsUp, 128, 64);
  pcd_final_kernel<<<cdiv(rowsUp, 256), 256, 0, stream>>>(ping, pcdW[1], pcdB[1],
                                                          (float*)d_out, rowsUp);
}
